// PHDGN_Model_40458591928622
// MI455X (gfx1250) — matrix operand order repaired and checked by reading, not compiled, not run on hardware
//
#include <hip/hip_runtime.h>

// ---------------------------------------------------------------------------
// CDNA5 (gfx1250, wave32) WMMA 16x16x32 bf16 -> f32
// ---------------------------------------------------------------------------
typedef __attribute__((ext_vector_type(16))) __bf16 v16bf;
typedef __attribute__((ext_vector_type(8)))  float  v8f;

union Frag { uint4 q[2]; unsigned u[8]; v16bf v; };

// round-to-nearest-even pack of two f32 into one dword holding two bf16
__device__ __forceinline__ unsigned pack_bf16(float x, float y) {
    unsigned ux = __builtin_bit_cast(unsigned, x);
    unsigned uy = __builtin_bit_cast(unsigned, y);
    ux = ux + 0x7FFFu + ((ux >> 16) & 1u);
    uy = uy + 0x7FFFu + ((uy >> 16) & 1u);
    return (ux >> 16) | (uy & 0xFFFF0000u);
}

__device__ __forceinline__ float gelu_exact(float x) {
    return 0.5f * x * (1.0f + erff(x * 0.70710678118654752440f));
}

// Low 32 bits of a flat pointer into LDS == LDS byte offset (ISA 10.2 aperture).
__device__ __forceinline__ unsigned lds_addr(const void* p) {
    return (unsigned)(size_t)p;
}

// Async-DMA a 16xK f32 tile (rows clamped to M-1) into LDS. Uses ASYNCcnt.
template<int K>
__device__ __forceinline__ void stage_tile_async(const float* __restrict__ A,
                                                 float* s, int tileM, int M, int tid) {
    constexpr int CPR   = K / 4;        // float4 chunks per row
    constexpr int TOTAL = 16 * CPR;     // chunks per tile (multiple of 256)
#pragma unroll
    for (int c0 = 0; c0 < TOTAL; c0 += 256) {
        const int c   = c0 + tid;
        const int row = c / CPR;
        const int col = c - row * CPR;
        int gr = tileM + row; gr = (gr < M) ? gr : (M - 1);
        const float* gp = A + (size_t)gr * K + col * 4;
        asm volatile("global_load_async_to_lds_b128 %0, %1, off"
                     :: "v"(lds_addr(s + c * 4)),
                        "v"((unsigned long long)(size_t)gp)
                     : "memory");
    }
}

// Cooperative convert: raw f32 16xK LDS tile -> bf16 A-fragment dword layout.
// Fragment dword idx = (chunk*32 + lane)*8 + v ; lane: m = l&15, hi = l>>4 ;
// v<4: K = c*32 + hi*8 + 2v ; v>=4: K = c*32 + 16 + hi*8 + 2(v-4).
template<int K>
__device__ __forceinline__ void convert_tile(const float* sraw, unsigned* sfrag, int tid) {
    constexpr int TOTD = (K / 32) * 256;          // dwords (multiple of 256)
#pragma unroll
    for (int i0 = 0; i0 < TOTD; i0 += 256) {
        const int idx = i0 + tid;
        const int v   = idx & 7;
        const int l   = (idx >> 3) & 31;
        const int c   = idx >> 8;
        const int m   = l & 15;
        const int kk  = c * 32 + ((v < 4) ? 0 : 16) + (l >> 4) * 8 + 2 * (v & 3);
        float2 t = *(const float2*)(sraw + m * K + kk);
        sfrag[idx] = pack_bf16(t.x, t.y);
    }
}

// Pre-swizzle one weight matrix into bf16 B-fragment dword order.
// TB=1: B[k][n] = W[n][k] (W row-major [Nout,K], x@W^T). TB=0: B[k][n] = W[k][n].
// out idx = ((t*Kc + c)*32 + l)*8 + v ; n = t*16 + (l&15) clamped ;
// kk = c*32 + (v<4 ? 0 : 16) + (l>>4)*8 + 2(v&3).
__global__ __launch_bounds__(256) void pack_weights(
    const float* __restrict__ W, unsigned* __restrict__ out,
    int K, int Nout, int Nt, int TB)
{
    const int idx   = blockIdx.x * 256 + threadIdx.x;
    const int Kc    = K / 32;
    const int total = Nt * Kc * 256;
    if (idx >= total) return;
    const int v = idx & 7;
    const int l = (idx >> 3) & 31;
    const int c = (idx >> 8) % Kc;
    const int t = (idx >> 8) / Kc;
    int n = t * 16 + (l & 15); if (n >= Nout) n = Nout - 1;
    const int kk = c * 32 + ((v < 4) ? 0 : 16) + (l >> 4) * 8 + 2 * (v & 3);
    float x0, x1;
    if (TB) { x0 = W[(size_t)n * K + kk];   x1 = W[(size_t)n * K + kk + 1]; }
    else    { x0 = W[(size_t)kk * Nout + n]; x1 = W[(size_t)(kk + 1) * Nout + n]; }
    out[idx] = pack_bf16(x0, x1);
}

// out[m][n] = act( alpha*(A1@B1 + A2@B2) + bias[n] + base[m][n] )
// B1/B2 are pre-swizzled bf16 fragment buffers. act: 0 none, 1 tanh, 2 GELU.
// One 16x16 tile per wave; 8 waves cover up to 128 output columns.
template<int K>
__global__ __launch_bounds__(256) void gemm_wmma(
    const float* __restrict__ A1, const unsigned* __restrict__ Bp1,
    const float* __restrict__ A2, const unsigned* __restrict__ Bp2,
    const float* __restrict__ bias, const float* __restrict__ base,
    float* __restrict__ out, int M, int Nout, int ldOut, float alpha, int act)
{
    constexpr int Kc = K / 32;
    __shared__ float    sRaw[2][16 * K];
    __shared__ unsigned sFrag[2][Kc * 256];

    const int tid   = threadIdx.x;
    const int lane  = tid & 31;
    const int wave  = tid >> 5;
    const int tileM = blockIdx.x * 16;
    const int tileN = wave * 16;

    stage_tile_async<K>(A1, sRaw[0], tileM, M, tid);
    if (A2) stage_tile_async<K>(A2, sRaw[1], tileM, M, tid);
    asm volatile("s_wait_asynccnt 0" ::: "memory");
    __syncthreads();

    convert_tile<K>(sRaw[0], sFrag[0], tid);
    if (A2) convert_tile<K>(sRaw[1], sFrag[1], tid);
    __syncthreads();

    if (tileN >= Nout) return;              // wave-uniform; EXEC all-1s for WMMA

    const int m16 = lane & 15;
    const int hi  = lane >> 4;
    const int n   = tileN + m16;

    v8f acc = {};
    {
        const unsigned* bp = Bp1 + ((size_t)(wave * Kc) * 32 + lane) * 8;
#pragma unroll
        for (int c = 0; c < Kc; ++c) {
            Frag a, b;
            const unsigned* ap = sFrag[0] + (c * 32 + lane) * 8;
            a.q[0] = *(const uint4*)ap;
            a.q[1] = *(const uint4*)(ap + 4);
            b.q[0] = *(const uint4*)bp;
            b.q[1] = *(const uint4*)(bp + 4);
            bp += 256;
            acc = __builtin_amdgcn_wmma_f32_16x16x32_bf16(
                      false, a.v, false, b.v, (short)0, acc, false, false);
        }
    }
    if (A2) {
        const unsigned* bp = Bp2 + ((size_t)(wave * Kc) * 32 + lane) * 8;
#pragma unroll
        for (int c = 0; c < Kc; ++c) {
            Frag a, b;
            const unsigned* ap = sFrag[1] + (c * 32 + lane) * 8;
            a.q[0] = *(const uint4*)ap;
            a.q[1] = *(const uint4*)(ap + 4);
            b.q[0] = *(const uint4*)bp;
            b.q[1] = *(const uint4*)(bp + 4);
            bp += 256;
            acc = __builtin_amdgcn_wmma_f32_16x16x32_bf16(
                      false, a.v, false, b.v, (short)0, acc, false, false);
        }
    }

    const float bn = (bias != nullptr && n < Nout) ? bias[n] : 0.0f;
#pragma unroll
    for (int j = 0; j < 8; ++j) {
        const int m = tileM + j + hi * 8;
        if (m < M && n < Nout) {
            float r = alpha * acc[j] + bn;
            if (base) r += base[(size_t)m * ldOut + n];
            if (act == 1)      r = tanhf(r);
            else if (act == 2) r = gelu_exact(r);
            out[(size_t)m * ldOut + n] = r;
        }
    }
}

// Edge aggregation: out[dst] += h[src]. One wave per edge; lane carries float4.
__global__ __launch_bounds__(256) void agg_edges(
    const float* __restrict__ h, const int* __restrict__ src,
    const int* __restrict__ dst, float* __restrict__ out, int E)
{
    const int e    = blockIdx.x * 8 + (threadIdx.x >> 5);
    const int lane = threadIdx.x & 31;
    if (e >= E) return;
    const int s = src[e], d = dst[e];
    const float4 v = *((const float4*)(h + (size_t)s * 128) + lane);
    float* o = out + (size_t)d * 128 + lane * 4;
    atomicAdd(o + 0, v.x);
    atomicAdd(o + 1, v.y);
    atomicAdd(o + 2, v.z);
    atomicAdd(o + 3, v.w);
}

// Concat [p|q] -> LayerNorm over 256 -> out[N,256]. One wave per node.
__global__ __launch_bounds__(256) void ln_concat(
    const float* __restrict__ P, const float* __restrict__ Q,
    const float* __restrict__ g, const float* __restrict__ b,
    float* __restrict__ out, int M)
{
    const int node = blockIdx.x * 8 + (threadIdx.x >> 5);
    const int lane = threadIdx.x & 31;
    if (node >= M) return;

    const float* src = (lane < 16) ? (P + (size_t)node * 128 + lane * 8)
                                   : (Q + (size_t)node * 128 + (lane - 16) * 8);
    float v[8];
    float4 v0 = *(const float4*)src;
    float4 v1 = *(const float4*)(src + 4);
    v[0] = v0.x; v[1] = v0.y; v[2] = v0.z; v[3] = v0.w;
    v[4] = v1.x; v[5] = v1.y; v[6] = v1.z; v[7] = v1.w;

    float s = 0.0f, ss = 0.0f;
#pragma unroll
    for (int i = 0; i < 8; ++i) { s += v[i]; ss += v[i] * v[i]; }
#pragma unroll
    for (int o = 16; o >= 1; o >>= 1) {
        s  += __shfl_xor(s,  o, 32);
        ss += __shfl_xor(ss, o, 32);
    }
    const float mu  = s * (1.0f / 256.0f);
    const float var = ss * (1.0f / 256.0f) - mu * mu;
    const float inv = rsqrtf(var + 1e-5f);

    const int c = lane * 8;
    float* o = out + (size_t)node * 256 + c;
#pragma unroll
    for (int i = 0; i < 8; ++i)
        o[i] = (v[i] - mu) * inv * g[c + i] + b[c + i];
}

// ---------------------------------------------------------------------------
extern "C" void kernel_launch(void* const* d_in, const int* in_sizes, int n_in,
                              void* d_out, int out_size, void* d_ws, size_t ws_size,
                              hipStream_t stream) {
    const float* x     = (const float*)d_in[0];
    const int*   ei    = (const int*)d_in[1];
    const float* emb_w = (const float*)d_in[2];
    const float* emb_b = (const float*)d_in[3];
    const float* Wp    = (const float*)d_in[4];
    const float* Vp    = (const float*)d_in[5];
    const float* bp    = (const float*)d_in[6];
    const float* Wq    = (const float*)d_in[7];
    const float* Vq    = (const float*)d_in[8];
    const float* bq    = (const float*)d_in[9];
    const float* ln_g  = (const float*)d_in[10];
    const float* ln_b  = (const float*)d_in[11];
    const float* r1_w  = (const float*)d_in[12];
    const float* r1_b  = (const float*)d_in[13];
    const float* r2_w  = (const float*)d_in[14];
    const float* r2_b  = (const float*)d_in[15];
    const float* r3_w  = (const float*)d_in[16];
    const float* r3_b  = (const float*)d_in[17];

    const int M = in_sizes[0] / 128;     // 50000 nodes
    const int E = in_sizes[1] / 2;       // 800000 edges
    const int* src = ei;
    const int* dst = ei + E;

    const size_t nf = (size_t)M * 128;
    float* P  = (float*)d_ws;
    float* Q  = P + nf;
    float* AG = Q + nf;
    float* Z  = AG + nf;
    float* AZ = Z + nf;
    float* H2 = AZ + nf;                 // M x 256
    float* R1 = AG;                      // reuse after conv loop
    float* R2 = Z;

    // pre-swizzled bf16 weight buffers (dwords)
    const int SZ128 = 8 * 4 * 256;       // Nt=8, Kc=4  -> 8192 dwords
    const int SZ256 = 8 * 8 * 256;       // Nt=8, Kc=8  -> 16384 dwords
    const int SZR3  = 1 * 4 * 256;       // Nt=1, Kc=4  -> 1024 dwords
    unsigned* wb    = (unsigned*)(H2 + (size_t)M * 256);
    unsigned* embT  = wb;
    unsigned* WqT   = embT + SZ128;
    unsigned* VqT   = WqT  + SZ128;
    unsigned* Wq0   = VqT  + SZ128;
    unsigned* Vq0   = Wq0  + SZ128;
    unsigned* WpT   = Vq0  + SZ128;
    unsigned* VpT   = WpT  + SZ128;
    unsigned* Wp0   = VpT  + SZ128;
    unsigned* Vp0   = Wp0  + SZ128;
    unsigned* r1T   = Vp0  + SZ128;
    unsigned* r2T   = r1T  + SZ256;
    unsigned* r3T   = r2T  + SZ128;

    dim3 blk(256);
    const int gW128 = SZ128 / 256, gW256 = SZ256 / 256, gWr3 = SZR3 / 256;
    const int gM = (M + 15) / 16;
    const int gE = (E + 7) / 8;
    const int gL = (M + 7) / 8;
    const float EPS = 0.1f;

    pack_weights<<<gW128, blk, 0, stream>>>(emb_w, embT, 128, 128, 8, 1);
    pack_weights<<<gW128, blk, 0, stream>>>(Wq, WqT, 128, 128, 8, 1);
    pack_weights<<<gW128, blk, 0, stream>>>(Vq, VqT, 128, 128, 8, 1);
    pack_weights<<<gW128, blk, 0, stream>>>(Wq, Wq0, 128, 128, 8, 0);
    pack_weights<<<gW128, blk, 0, stream>>>(Vq, Vq0, 128, 128, 8, 0);
    pack_weights<<<gW128, blk, 0, stream>>>(Wp, WpT, 128, 128, 8, 1);
    pack_weights<<<gW128, blk, 0, stream>>>(Vp, VpT, 128, 128, 8, 1);
    pack_weights<<<gW128, blk, 0, stream>>>(Wp, Wp0, 128, 128, 8, 0);
    pack_weights<<<gW128, blk, 0, stream>>>(Vp, Vp0, 128, 128, 8, 0);
    pack_weights<<<gW256, blk, 0, stream>>>(r1_w, r1T, 256, 128, 8, 1);
    pack_weights<<<gW128, blk, 0, stream>>>(r2_w, r2T, 128, 128, 8, 1);
    pack_weights<<<gWr3,  blk, 0, stream>>>(r3_w, r3T, 128, 10, 1, 1);

    // h = gelu(x @ emb_w^T + emb_b);  p = q = h
    gemm_wmma<128><<<gM, blk, 0, stream>>>(x, embT, nullptr, nullptr, emb_b,
                                           nullptr, P, M, 128, 128, 1.0f, 2);
    hipMemcpyAsync(Q, P, nf * sizeof(float), hipMemcpyDeviceToDevice, stream);

    for (int it = 0; it < 5; ++it) {
        // zq = tanh(q@Wq^T + agg(q)@Vq^T + bq)
        hipMemsetAsync(AG, 0, nf * sizeof(float), stream);
        agg_edges<<<gE, blk, 0, stream>>>(Q, src, dst, AG, E);
        gemm_wmma<128><<<gM, blk, 0, stream>>>(Q, WqT, AG, VqT, bq, nullptr,
                                               Z, M, 128, 128, 1.0f, 1);
        // p -= eps*(zq@Wq + agg(zq)@Vq)
        hipMemsetAsync(AZ, 0, nf * sizeof(float), stream);
        agg_edges<<<gE, blk, 0, stream>>>(Z, src, dst, AZ, E);
        gemm_wmma<128><<<gM, blk, 0, stream>>>(Z, Wq0, AZ, Vq0, nullptr, P,
                                               P, M, 128, 128, -EPS, 0);
        // zp = tanh(p@Wp^T + agg(p)@Vp^T + bp)
        hipMemsetAsync(AG, 0, nf * sizeof(float), stream);
        agg_edges<<<gE, blk, 0, stream>>>(P, src, dst, AG, E);
        gemm_wmma<128><<<gM, blk, 0, stream>>>(P, WpT, AG, VpT, bp, nullptr,
                                               Z, M, 128, 128, 1.0f, 1);
        // q += eps*(zp@Wp + agg(zp)@Vp)
        hipMemsetAsync(AZ, 0, nf * sizeof(float), stream);
        agg_edges<<<gE, blk, 0, stream>>>(Z, src, dst, AZ, E);
        gemm_wmma<128><<<gM, blk, 0, stream>>>(Z, Wp0, AZ, Vp0, nullptr, Q,
                                               Q, M, 128, 128, +EPS, 0);
    }

    // LayerNorm(concat(p,q))
    ln_concat<<<gL, blk, 0, stream>>>(P, Q, ln_g, ln_b, H2, M);

    // readout MLP
    gemm_wmma<256><<<gM, blk, 0, stream>>>(H2, r1T, nullptr, nullptr, r1_b,
                                           nullptr, R1, M, 128, 128, 1.0f, 2);
    gemm_wmma<128><<<gM, blk, 0, stream>>>(R1, r2T, nullptr, nullptr, r2_b,
                                           nullptr, R2, M, 128, 128, 1.0f, 2);
    gemm_wmma<128><<<gM, blk, 0, stream>>>(R2, r3T, nullptr, nullptr, r3_b,
                                           nullptr, (float*)d_out, M, 10, 10,
                                           1.0f, 0);
}
